// simpleDFN_41790031790110
// MI455X (gfx1250) — matrix operand order repaired and checked by reading, not compiled, not run on hardware
//
#include <hip/hip_runtime.h>

typedef _Float16 h8  __attribute__((ext_vector_type(8)));
typedef _Float16 h16 __attribute__((ext_vector_type(16)));
typedef float    f8v __attribute__((ext_vector_type(8)));
typedef float    f4v __attribute__((ext_vector_type(4)));

// N=16, C=128, H=W=96, OUT_C=128, padded result 98x98 NHWC f16.

// ---------------- zero the padded borders of the NHWC f16 result ----------------
__global__ void k_zero_border(_Float16* __restrict__ resP) {
    int idx = blockIdx.x * 256 + threadIdx.x;              // 16 * 388 * 128 = 794624
    const int per_n = 388 * 128;
    if (idx >= 16 * per_n) return;
    int n = idx / per_n;
    int r = idx % per_n;
    int cell = r >> 7;
    int c    = r & 127;
    int y, xp;
    if      (cell <  98) { y = 0;              xp = cell;       }
    else if (cell < 196) { y = 97;             xp = cell - 98;  }
    else if (cell < 292) { y = cell - 196 + 1; xp = 0;          }
    else                 { y = cell - 292 + 1; xp = 97;         }
    resP[(((size_t)n * 98 + y) * 98 + xp) * 128 + c] = (_Float16)0.f;
}

// ---------------- adaptive avg pool 96x96 -> 3x3 (exact 32x32 block means) ----------------
__global__ void k_pool(const float* __restrict__ y, float* __restrict__ pooled) {
    __shared__ float red[256];
    int nc  = blockIdx.x;                   // n*128 + c   (2048 blocks)
    int tid = threadIdx.x;
    const float* yp = y + (size_t)nc * 96 * 96;
    for (int rgn = 0; rgn < 9; ++rgn) {
        int bi = rgn / 3, bj = rgn % 3;
        float acc = 0.f;
        for (int t = tid; t < 1024; t += 256) {
            int hh = bi * 32 + (t >> 5);
            int ww = bj * 32 + (t & 31);
            acc += yp[hh * 96 + ww];
        }
        red[tid] = acc; __syncthreads();
        for (int off = 128; off > 0; off >>= 1) {
            if (tid < off) red[tid] += red[tid + off];
            __syncthreads();
        }
        if (tid == 0) pooled[(size_t)nc * 9 + rgn] = red[0] * (1.f / 1024.f);
        __syncthreads();
    }
}

// ---------------- 1x1 channel mix: k[n,o,ij] = sum_c wk[o,c]*pooled[n,c,ij] + bk[o] ----------------
__global__ void k_mix(const float* __restrict__ wk, const float* __restrict__ bk,
                      const float* __restrict__ pooled, float* __restrict__ kbuf) {
    int idx = blockIdx.x * 256 + threadIdx.x;              // 16*128*9 = 18432
    if (idx >= 16 * 128 * 9) return;
    int ij = idx % 9;
    int o  = (idx / 9) % 128;
    int n  = idx / (9 * 128);
    float s = bk[o];
    const float* wr = wk + (size_t)o * 128;
    const float* pr = pooled + (size_t)n * 128 * 9 + ij;
    #pragma unroll 8
    for (int c = 0; c < 128; ++c) s += wr[c] * pr[(size_t)c * 9];
    kbuf[((size_t)n * 128 + o) * 9 + ij] = s;
}

// ---------------- pack wf (fp32 OIHW) into per-lane f16 B-fragments ----------------
// bpack[((tap*4+ck)*8+nt)*512 + lane*16 + h]
//   lane<16 : (K=(h>>3)*16+(h&7),   N=lane   ) -> c = ck*32 + K, o = nt*16 + lane
//   lane>=16: (K=(h>>3)*16+8+(h&7), N=lane-16) -> c = ck*32 + K, o = nt*16 + lane-16
__global__ void k_pack(const float* __restrict__ wf, _Float16* __restrict__ bpack) {
    int idx = blockIdx.x * 256 + threadIdx.x;              // 9*4*8*32*16 = 147456
    if (idx >= 147456) return;
    int hh   = idx & 15;
    int lane = (idx >> 4) & 31;
    int nt   = (idx >> 9) & 7;
    int ck   = (idx >> 12) & 3;
    int tap  = idx >> 14;
    int dy = tap / 3, dx = tap % 3;
    int o = nt * 16 + (lane & 15);
    int c = ck * 32 + ((hh >> 3) << 4) + ((lane >> 4) << 3) + (hh & 7);
    bpack[idx] = (_Float16)wf[(((size_t)o * 128 + c) * 3 + dy) * 3 + dx];
}

// ---------------- depthwise 3x3 with per-(n,c) constant kernel; NCHW f32 -> padded NHWC f16 ----------------
__global__ void k_dw(const float* __restrict__ x, const float* __restrict__ kbuf,
                     _Float16* __restrict__ resP) {
    __shared__ float xs[32 * 3 * 100];
    __shared__ float kk[32 * 9];
    int n = blockIdx.x / 96;             // 1536 blocks = N*H
    int h = blockIdx.x % 96;
    int tid = threadIdx.x;
    for (int cc = 0; cc < 4; ++cc) {
        int cbase = cc * 32;
        for (int i = tid; i < 32 * 9; i += 256)
            kk[i] = kbuf[((size_t)n * 128 + cbase + i / 9) * 9 + (i % 9)];
        for (int i = tid; i < 32 * 3 * 98; i += 256) {
            int ci  = i / (3 * 98);
            int rem = i % (3 * 98);
            int r   = rem / 98;
            int wp  = rem % 98;
            int hy = h + r - 1;
            int wx = wp - 1;
            float v = 0.f;
            if (hy >= 0 && hy < 96 && wx >= 0 && wx < 96)
                v = x[(((size_t)n * 128 + cbase + ci) * 96 + hy) * 96 + wx];
            xs[(ci * 3 + r) * 100 + wp] = v;
        }
        __syncthreads();
        for (int i = tid; i < 96 * 32; i += 256) {
            int ci = i & 31;
            int w  = i >> 5;
            const float* kc = kk + ci * 9;
            const float* xr = xs + ci * 3 * 100;
            float s = 0.f;
            #pragma unroll
            for (int r = 0; r < 3; ++r)
                #pragma unroll
                for (int d = 0; d < 3; ++d)
                    s += xr[r * 100 + w + d] * kc[r * 3 + d];
            resP[(((size_t)n * 98 + h + 1) * 98 + (w + 1)) * 128 + cbase + ci] = (_Float16)s;
        }
        __syncthreads();
    }
}

// ---------------- helpers ----------------
__device__ __forceinline__ h16 ld_a(const _Float16* p) {     // A frag: halves at +0 and +16
    h8 lo = *(const h8*)p;
    h8 hi = *(const h8*)(p + 16);
    return __builtin_shufflevector(lo, hi, 0,1,2,3,4,5,6,7,8,9,10,11,12,13,14,15);
}
__device__ __forceinline__ h16 ld_b(const _Float16* p) {     // B frag: 16 contiguous halves
    h8 lo = *(const h8*)p;
    h8 hi = *(const h8*)(p + 8);
    return __builtin_shufflevector(lo, hi, 0,1,2,3,4,5,6,7,8,9,10,11,12,13,14,15);
}

// ---------------- final conv as implicit GEMM via v_wmma_f32_16x16x32_f16 ----------------
// Per wave: 2 M-tiles (32 consecutive w) x 4 N-tiles (64 out channels).
// B slab for the current tap (32KB) is staged into LDS with async global->LDS copies.
__global__ void __launch_bounds__(256) k_gemm(const _Float16* __restrict__ resP,
                                              const _Float16* __restrict__ bpack,
                                              const float* __restrict__ bf,
                                              float* __restrict__ out) {
    __shared__ _Float16 Bsh[16384];      // 32KB: [ck 4][ntile 8][lane 32][16]
    int tid  = threadIdx.x;
    int lane = tid & 31;
    int wave = tid >> 5;
    int job  = blockIdx.x * 8 + wave;    // 9216 jobs, 1152 blocks
    int nh   = job & 1;                  // N half: o base = nh*64
    int mp   = job >> 1;                 // 0..4607
    int n    = mp / 288;                 // 288 = 3*96
    int rem  = mp % 288;
    int h    = rem / 3;
    int w0   = (rem % 3) * 32;
    int m    = lane & 15;
    int hi   = lane >> 4;

    unsigned lds0;
    {
        auto p3 = (__attribute__((address_space(3))) _Float16*)Bsh;
        lds0 = (unsigned)(unsigned long long)p3;
    }

    f8v acc[2][4];
    #pragma unroll
    for (int mt = 0; mt < 2; ++mt)
        #pragma unroll
        for (int nt = 0; nt < 4; ++nt)
            acc[mt][nt] = (f8v){0.f, 0.f, 0.f, 0.f, 0.f, 0.f, 0.f, 0.f};

    #pragma unroll 1
    for (int tap = 0; tap < 9; ++tap) {
        __syncthreads();                 // previous tap's LDS reads complete
        // ---- async stage this tap's 32KB B slab into LDS (ASYNCcnt path) ----
        {
            const _Float16* gsrc = bpack + (size_t)tap * 16384;
            #pragma unroll
            for (int i = 0; i < 8; ++i) {
                unsigned off = (unsigned)(tid + 256 * i) * 16u;
                asm volatile("global_load_async_to_lds_b128 %0, %1, %2"
                             :: "v"(lds0 + off), "v"(off), "s"(gsrc)
                             : "memory");
            }
            asm volatile("s_wait_asynccnt 0x0" ::: "memory");
        }
        __syncthreads();

        int dy = tap / 3, dx = tap % 3;
        const _Float16* ap0 = resP + (((size_t)n * 98 + h + dy) * 98 + (w0 + m + dx)) * 128 + hi * 8;
        const _Float16* ap1 = ap0 + 16 * 128;
        const _Float16* bl  = Bsh + (size_t)(nh * 4) * 512 + (size_t)lane * 16;

        #pragma unroll
        for (int ck = 0; ck < 4; ++ck) {
            // hoist all loads for this chunk: one wait covers 8 WMMAs
            h16 a0 = ld_a(ap0 + ck * 32);
            h16 a1 = ld_a(ap1 + ck * 32);
            h16 bfr[4];
            #pragma unroll
            for (int nt = 0; nt < 4; ++nt)
                bfr[nt] = ld_b(bl + (size_t)(ck * 8 + nt) * 512);
            #pragma unroll
            for (int nt = 0; nt < 4; ++nt) {
                acc[0][nt] = __builtin_amdgcn_wmma_f32_16x16x32_f16(
                    false, a0, false, bfr[nt], (short)0, acc[0][nt], false, false);
                acc[1][nt] = __builtin_amdgcn_wmma_f32_16x16x32_f16(
                    false, a1, false, bfr[nt], (short)0, acc[1][nt], false, false);
            }
        }
    }

    // Store: D layout — VGPR r: lane<16 -> (M=r, N=lane); lane>=16 -> (M=r+8, N=lane-16)
    int xoff = hi * 8;
    #pragma unroll
    for (int nt = 0; nt < 4; ++nt) {
        int o = nh * 64 + nt * 16 + m;
        float bias = bf[o];
        #pragma unroll
        for (int mt = 0; mt < 2; ++mt) {
            float* op = out + (((size_t)n * 128 + o) * 96 + h) * 96 + (w0 + mt * 16) + xoff;
            f4v v0 = { acc[mt][nt][0] + bias, acc[mt][nt][1] + bias,
                       acc[mt][nt][2] + bias, acc[mt][nt][3] + bias };
            f4v v1 = { acc[mt][nt][4] + bias, acc[mt][nt][5] + bias,
                       acc[mt][nt][6] + bias, acc[mt][nt][7] + bias };
            *(f4v*)op       = v0;
            *(f4v*)(op + 4) = v1;
        }
    }
}

extern "C" void kernel_launch(void* const* d_in, const int* in_sizes, int n_in,
                              void* d_out, int out_size, void* d_ws, size_t ws_size,
                              hipStream_t stream) {
    (void)in_sizes; (void)n_in; (void)out_size; (void)ws_size;
    const float* x  = (const float*)d_in[0];
    const float* y  = (const float*)d_in[1];
    const float* wk = (const float*)d_in[2];
    const float* bk = (const float*)d_in[3];
    const float* wf = (const float*)d_in[4];
    const float* bf = (const float*)d_in[5];
    float* out = (float*)d_out;

    // Workspace layout (256B aligned):
    //   resP  : 16*98*98*128 f16 = 39,337,984 B @ 0
    //   pooled: 16*128*9 f32     =     73,728 B @ 39,337,984
    //   kbuf  : 16*128*9 f32     =     73,728 B @ 39,411,712
    //   bpack : 9*4*8*512 f16    =    294,912 B @ 39,485,440   (total ~39.8 MB)
    char* ws = (char*)d_ws;
    _Float16* resP   = (_Float16*)(ws);
    float*    pooled = (float*)(ws + 39337984);
    float*    kbuf   = (float*)(ws + 39411712);
    _Float16* bpack  = (_Float16*)(ws + 39485440);

    k_zero_border<<<3104, 256, 0, stream>>>(resP);
    k_pool       <<<2048, 256, 0, stream>>>(y, pooled);
    k_mix        <<<  72, 256, 0, stream>>>(wk, bk, pooled, kbuf);
    k_pack       <<< 576, 256, 0, stream>>>(wf, bpack);
    k_dw         <<<1536, 256, 0, stream>>>(x, kbuf, resP);
    k_gemm       <<<1152, 256, 0, stream>>>(resP, bpack, bf, out);
}
